// GET_inter_79766132622008
// MI455X (gfx1250) — matrix operand order repaired and checked by reading, not compiled, not run on hardware
//
#include <hip/hip_runtime.h>
#include <hip/hip_bf16.h>

// Problem constants (fixed by the reference harness)
#define N_DSTC 6144
#define N_SRCC 384
#define N_ALLC (N_DSTC + N_SRCC)   // 6528
#define DD     256
#define DQC    16
#define E_BGC  196608
#define E_TGC  (N_DSTC * 64)       // 393216

typedef __attribute__((ext_vector_type(16))) _Float16 v16h;
typedef __attribute__((ext_vector_type(8)))  _Float16 v8h;
typedef __attribute__((ext_vector_type(8)))  float    v8f;

// ---------------------------------------------------------------------------
// Utility: zero a float region
// ---------------------------------------------------------------------------
__global__ void zero_kernel(float* __restrict__ p, int n) {
    int i = blockIdx.x * blockDim.x + threadIdx.x;
    if (i < n) p[i] = 0.f;
}

// ---------------------------------------------------------------------------
// bg preprocessing pass 1: per-dst edge count and sum of E (E > 0 only)
// ---------------------------------------------------------------------------
__global__ void bg_pass1_kernel(const float* __restrict__ bgE,
                                const int* __restrict__ bg_dst,
                                float* __restrict__ cnt, float* __restrict__ sumE) {
    int e = blockIdx.x * blockDim.x + threadIdx.x;
    if (e >= E_BGC) return;
    float Ev = bgE[e];
    if (Ev > 0.f) {
        int d = bg_dst[e];
        atomicAdd(&cnt[d], 1.f);
        atomicAdd(&sumE[d], Ev);
    }
}

__global__ void div_kernel(const float* __restrict__ sumE, const float* __restrict__ cnt,
                           float* __restrict__ Div) {
    int d = blockIdx.x * blockDim.x + threadIdx.x;
    if (d >= N_DSTC) return;
    Div[d] = sumE[d] / fmaxf(cnt[d], 1.f);
}

__global__ void bg_pass2_kernel(const float* __restrict__ bgE,
                                const int* __restrict__ bg_src, const int* __restrict__ bg_dst,
                                const float* __restrict__ Div, float* __restrict__ prsum) {
    int e = blockIdx.x * blockDim.x + threadIdx.x;
    if (e >= E_BGC) return;
    float Ev = bgE[e];
    if (Ev > 0.f) {
        float det = Div[bg_src[e]] - Div[bg_dst[e]];
        if (det > 0.f) atomicAdd(&prsum[bg_dst[e]], det);
    }
}

__global__ void pr_kernel(const float* __restrict__ prsum, const float* __restrict__ cnt,
                          float* __restrict__ pr) {
    int d = blockIdx.x * blockDim.x + threadIdx.x;
    if (d >= N_DSTC) return;
    pr[d] = prsum[d] / fmaxf(cnt[d], 1.f);
}

// ---------------------------------------------------------------------------
// Exact top-k: 384 smallest pr, ties broken by lower index (matches
// jax.lax.top_k(-pr, 384)). Bitonic sort of 8192 unique 64-bit keys in LDS.
// key = (monotone_u32(pr) << 32) | index  -> ascending sort.
// ---------------------------------------------------------------------------
__global__ void __launch_bounds__(1024) topk_kernel(const float* __restrict__ pr,
                                                    int* __restrict__ ids) {
    __shared__ unsigned long long key[8192];   // 64 KB LDS (<= 320 KB/WGP)
    int t = threadIdx.x;
    for (int i = t; i < 8192; i += 1024) {
        unsigned long long k;
        if (i < N_DSTC) {
            unsigned u = __float_as_uint(pr[i]);
            u = (u & 0x80000000u) ? ~u : (u | 0x80000000u);  // total order on floats
            k = ((unsigned long long)u << 32) | (unsigned)i;
        } else {
            k = ~0ull;
        }
        key[i] = k;
    }
    __syncthreads();
    for (int size = 2; size <= 8192; size <<= 1) {
        for (int stride = size >> 1; stride >= 1; stride >>= 1) {
            for (int i = t; i < 4096; i += 1024) {
                int blk = i / stride;
                int pos = i - blk * stride;
                int idx = blk * 2 * stride + pos;
                int par = idx + stride;
                unsigned long long a = key[idx], b = key[par];
                bool asc = ((idx & size) == 0);
                if ((a > b) == asc) { key[idx] = b; key[par] = a; }
            }
            __syncthreads();
        }
    }
    for (int i = t; i < N_SRCC; i += 1024) ids[i] = (int)(key[i] & 0xFFFFFFFFu);
}

// B^T f16 for GEMM1: Bt1[n][k] = Xe[src_ids[k]][n]   (256 x 384 halves)
__global__ void gather_bt1_kernel(const float* __restrict__ Xe, const int* __restrict__ ids,
                                  _Float16* __restrict__ Bt1) {
    int i = blockIdx.x * blockDim.x + threadIdx.x;
    if (i >= DD * N_SRCC) return;
    int n = i / N_SRCC, k = i - n * N_SRCC;
    Bt1[i] = (_Float16)Xe[(size_t)ids[k] * DD + n];
}

// Transpose+convert weight: Wt[n][k] = W[k][n]  (256 x 256 halves)
__global__ void conv_wt_kernel(const float* __restrict__ W, _Float16* __restrict__ Wt) {
    int i = blockIdx.x * blockDim.x + threadIdx.x;
    if (i >= DD * DD) return;
    int n = i >> 8, k = i & (DD - 1);
    Wt[i] = (_Float16)W[(size_t)k * DD + n];
}

// Flat f32 -> f16 convert (for Adense after atomic accumulation)
__global__ void conv_f16_kernel(const float* __restrict__ src, _Float16* __restrict__ dst, int n) {
    int i = blockIdx.x * blockDim.x + threadIdx.x;
    if (i < n) dst[i] = (_Float16)src[i];
}

// q = Xe_all @ W_dis * 0.25  (6528 x 16, exact f32; SRC rows gathered via ids)
__global__ void qproj_kernel(const float* __restrict__ Xe, const int* __restrict__ ids,
                             const float* __restrict__ Wdis, float* __restrict__ q) {
    int tid = blockIdx.x * blockDim.x + threadIdx.x;
    if (tid >= N_ALLC * DQC) return;
    int row = tid >> 4, j = tid & 15;
    const float* x = (row < N_DSTC) ? (Xe + (size_t)row * DD)
                                    : (Xe + (size_t)ids[row - N_DSTC] * DD);
    float acc = 0.f;
    for (int k = 0; k < DD; ++k) acc += x[k] * Wdis[k * DQC + j];
    q[tid] = acc * 0.25f;
}

// Per-edge attention scores: scatter aij into dense A, accumulate A_sum/R_sum.
__global__ void attn_edge_kernel(const float* __restrict__ q,
                                 const int* __restrict__ tg_src, const int* __restrict__ tg_dst,
                                 float* __restrict__ Adense,
                                 float* __restrict__ Asum, float* __restrict__ Rsum) {
    int e = blockIdx.x * blockDim.x + threadIdx.x;
    if (e >= E_TGC) return;
    int sg = tg_src[e];
    int d  = tg_dst[e];
    const float* qs = q + (size_t)sg * DQC;
    const float* qd = q + (size_t)d  * DQC;
    float dot = 0.f;
#pragma unroll
    for (int j = 0; j < DQC; ++j) dot += qs[j] * qd[j];
    if (dot > 0.f) {
        atomicAdd(&Adense[(size_t)d * N_SRCC + (sg - N_DSTC)], dot);
        atomicAdd(&Asum[d], dot);
    } else if (dot < 0.f) {
        atomicAdd(&Rsum[d], -dot);
    }
}

__global__ void denom_kernel(const float* __restrict__ Asum, const float* __restrict__ Rsum,
                             float* __restrict__ invden) {
    int d = blockIdx.x * blockDim.x + threadIdx.x;
    if (d >= N_DSTC) return;
    invden[d] = 1.f / (Asum[d] + Rsum[d] + 0.01f);
}

// ---------------------------------------------------------------------------
// WMMA GEMM: C[M,256] = A16[M,K] @ Bt16[256,K]^T (+ epilogue + BN col stats)
//   mode 0: C = (acc + Xe*Rsum[m]) * invden[m]          (Xe_trans)
//   mode 1: C = acc + bias[n]                           (FFN pre-BN)
// One wave per 16x16 tile, v_wmma_f32_16x16x32_f16, f32 accumulate.
// Operands pre-converted to f16; A row-major, B transposed, so every lane's
// fragment is contiguous 16-byte runs -> 4x global_load_b128 per WMMA.
// VGPR layouts per CDNA5 ISA 7.12.2 (wave32):
//   A: lane->M (lane&15), K packed 2/reg, half-split {0..7|8..15},{16..23|24..31}
//   B: lane->N (lane&15), K packed 2/reg, half-split {0..7|8..15},{16..23|24..31}
//   C: lane->N, reg j -> M = half*8 + j
// ---------------------------------------------------------------------------
__global__ void __launch_bounds__(256) wmma_gemm_kernel(
    const _Float16* __restrict__ A16, const _Float16* __restrict__ Bt16,
    float* __restrict__ C, int M, int K, int mode,
    const float* __restrict__ Xe, const float* __restrict__ Rsum,
    const float* __restrict__ invden, const float* __restrict__ bias,
    float* __restrict__ colsum, float* __restrict__ colsq)
{
    int wave = (blockIdx.x * blockDim.x + threadIdx.x) >> 5;
    int lane = threadIdx.x & 31;
    const int NT = DD / 16;                 // 16 tiles along N
    int tm = wave / NT, tn = wave - tm * NT;
    if (tm * 16 >= M) return;               // wave-uniform guard (EXEC stays all-1)
    int m0 = tm * 16, n0 = tn * 16;
    int half = lane >> 4, lr = lane & 15;

    v8f acc = {0.f, 0.f, 0.f, 0.f, 0.f, 0.f, 0.f, 0.f};
    const _Float16* arow = A16 + (size_t)(m0 + lr) * K;
    const _Float16* brow = Bt16 + (size_t)(n0 + lr) * K;
#pragma unroll 2
    for (int kk = 0; kk < K; kk += 32) {
        int ka = kk + half * 8;             // A regs 0..3
        int kc = kk + 16 + half * 8;        // A regs 4..7
        int kb = kk + half * 8;             // B regs 0..7
        v8h a0 = *(const v8h*)(arow + ka);  // 16B contiguous
        v8h a1 = *(const v8h*)(arow + kc);
        v8h b0 = *(const v8h*)(brow + kb);
        v8h b1 = *(const v8h*)(brow + kb + 16);
        v16h av = __builtin_shufflevector(a0, a1, 0,1,2,3,4,5,6,7,8,9,10,11,12,13,14,15);
        v16h bv = __builtin_shufflevector(b0, b1, 0,1,2,3,4,5,6,7,8,9,10,11,12,13,14,15);
        acc = __builtin_amdgcn_wmma_f32_16x16x32_f16(
            false, av, false, bv, (short)0, acc, false, false);
    }

    int n = n0 + lr;
    float s = 0.f, sq = 0.f;
#pragma unroll
    for (int j = 0; j < 8; ++j) {
        int m = m0 + half * 8 + j;
        float v = acc[j];
        if (mode == 0) v = (v + Xe[(size_t)m * DD + n] * Rsum[m]) * invden[m];
        else           v = v + bias[n];
        C[(size_t)m * DD + n] = v;
        s += v; sq += v * v;
    }
    atomicAdd(&colsum[n], s);
    atomicAdd(&colsq[n], sq);
}

// BN stats finalize (biased variance over batch of N_DST rows)
__global__ void bn_finalize_kernel(const float* __restrict__ colsum,
                                   const float* __restrict__ colsq,
                                   float* __restrict__ mean, float* __restrict__ istd) {
    int c = threadIdx.x;
    float m = colsum[c] * (1.0f / N_DSTC);
    float v = colsq[c] * (1.0f / N_DSTC) - m * m;
    mean[c] = m;
    istd[c] = rsqrtf(v + 1e-5f);
}

// Xe2 = Xe + BN1(Xe_trans); also emit f16 copy (A operand of GEMM2)
__global__ void ew_xe2_kernel(const float* __restrict__ Xe, const float* __restrict__ Xetr,
                              const float* __restrict__ mean, const float* __restrict__ istd,
                              const float* __restrict__ g, const float* __restrict__ b,
                              float* __restrict__ Xe2, _Float16* __restrict__ Xe2h) {
    int i = blockIdx.x * blockDim.x + threadIdx.x;
    if (i >= N_DSTC * DD) return;
    int c = i & (DD - 1);
    float v = Xe[i] + (Xetr[i] - mean[c]) * istd[c] * g[c] + b[c];
    Xe2[i]  = v;
    Xe2h[i] = (_Float16)v;
}

// h = relu(BN2(Z)) -> f16 only (sole consumer is GEMM3's A operand)
__global__ void ew_h_kernel(const float* __restrict__ Z,
                            const float* __restrict__ mean, const float* __restrict__ istd,
                            const float* __restrict__ g, const float* __restrict__ b,
                            _Float16* __restrict__ h16) {
    int i = blockIdx.x * blockDim.x + threadIdx.x;
    if (i >= N_DSTC * DD) return;
    int c = i & (DD - 1);
    float v = (Z[i] - mean[c]) * istd[c] * g[c] + b[c];
    h16[i] = (_Float16)(v > 0.f ? v : 0.f);
}

// out = Xe2 + BN3(Z)
__global__ void ew_out_kernel(const float* __restrict__ Xe2, const float* __restrict__ Z,
                              const float* __restrict__ mean, const float* __restrict__ istd,
                              const float* __restrict__ g, const float* __restrict__ b,
                              float* __restrict__ out) {
    int i = blockIdx.x * blockDim.x + threadIdx.x;
    if (i >= N_DSTC * DD) return;
    int c = i & (DD - 1);
    out[i] = Xe2[i] + (Z[i] - mean[c]) * istd[c] * g[c] + b[c];
}

// ---------------------------------------------------------------------------
extern "C" void kernel_launch(void* const* d_in, const int* in_sizes, int n_in,
                              void* d_out, int out_size, void* d_ws, size_t ws_size,
                              hipStream_t stream) {
    const float* Xe   = (const float*)d_in[0];
    const float* bgE  = (const float*)d_in[1];
    const float* Wdis = (const float*)d_in[2];
    const float* bn1g = (const float*)d_in[3];
    const float* bn1b = (const float*)d_in[4];
    const float* W1   = (const float*)d_in[5];
    const float* b1   = (const float*)d_in[6];
    const float* bn2g = (const float*)d_in[7];
    const float* bn2b = (const float*)d_in[8];
    const float* W2   = (const float*)d_in[9];
    const float* b2   = (const float*)d_in[10];
    const float* bn3g = (const float*)d_in[11];
    const float* bn3b = (const float*)d_in[12];
    const int* bg_src = (const int*)d_in[13];
    const int* bg_dst = (const int*)d_in[14];
    const int* tg_src = (const int*)d_in[15];
    const int* tg_dst = (const int*)d_in[16];
    float* out = (float*)d_out;
    (void)in_sizes; (void)n_in; (void)out_size; (void)ws_size;

    // Workspace layout (float units; all regions multiples of 8 floats -> 32B
    // aligned, so f16 b128 loads are aligned). Zeroed region first.
    float* ws = (float*)d_ws;
    size_t off = 0;
    float* cnt    = ws + off; off += N_DSTC;
    float* sumE   = ws + off; off += N_DSTC;
    float* prsum  = ws + off; off += N_DSTC;
    float* Asum   = ws + off; off += N_DSTC;
    float* Rsum   = ws + off; off += N_DSTC;
    float* cs1    = ws + off; off += DD;
    float* cq1    = ws + off; off += DD;
    float* cs2    = ws + off; off += DD;
    float* cq2    = ws + off; off += DD;
    float* cs3    = ws + off; off += DD;
    float* cq3    = ws + off; off += DD;
    float* Adense = ws + off; off += (size_t)N_DSTC * N_SRCC;
    int zero_count = (int)off;             // 2,391,552 floats
    float* Div    = ws + off; off += N_DSTC;
    float* pr     = ws + off; off += N_DSTC;
    float* invden = ws + off; off += N_DSTC;
    float* q      = ws + off; off += (size_t)N_ALLC * DQC;
    float* Xetr   = ws + off; off += (size_t)N_DSTC * DD;
    float* Xe2    = ws + off; off += (size_t)N_DSTC * DD;
    float* Z      = ws + off; off += (size_t)N_DSTC * DD;   // shared Z2/Z3
    float* mean1  = ws + off; off += DD;
    float* istd1  = ws + off; off += DD;
    float* mean2  = ws + off; off += DD;
    float* istd2  = ws + off; off += DD;
    float* mean3  = ws + off; off += DD;
    float* istd3  = ws + off; off += DD;
    // f16 regions (sizes counted in floats = halves/2)
    _Float16* Ah   = (_Float16*)(ws + off); off += (size_t)N_DSTC * N_SRCC / 2; // A of GEMM1
    _Float16* Bt1  = (_Float16*)(ws + off); off += (size_t)DD * N_SRCC / 2;     // B^T GEMM1
    _Float16* W1t  = (_Float16*)(ws + off); off += (size_t)DD * DD / 2;         // B^T GEMM2
    _Float16* W2t  = (_Float16*)(ws + off); off += (size_t)DD * DD / 2;         // B^T GEMM3
    _Float16* Xe2h = (_Float16*)(ws + off); off += (size_t)N_DSTC * DD / 2;     // A GEMM2
    _Float16* h16  = (_Float16*)(ws + off); off += (size_t)N_DSTC * DD / 2;     // A GEMM3
    int* src_ids   = (int*)(ws + off); off += N_SRCC;
    // total ~10.1M floats (~40 MB)

    const int T = 256;
    // 0) zero atomic accumulators
    zero_kernel<<<(zero_count + T - 1) / T, T, 0, stream>>>(ws, zero_count);

    // 1) bg preprocessing -> pr
    bg_pass1_kernel<<<E_BGC / T, T, 0, stream>>>(bgE, bg_dst, cnt, sumE);
    div_kernel<<<N_DSTC / T, T, 0, stream>>>(sumE, cnt, Div);
    bg_pass2_kernel<<<E_BGC / T, T, 0, stream>>>(bgE, bg_src, bg_dst, Div, prsum);
    pr_kernel<<<N_DSTC / T, T, 0, stream>>>(prsum, cnt, pr);

    // 2) exact stable top-k(384 smallest)
    topk_kernel<<<1, 1024, 0, stream>>>(pr, src_ids);

    // 3) operand prep: B^T f16 for all three GEMMs, q projection
    gather_bt1_kernel<<<(DD * N_SRCC) / T, T, 0, stream>>>(Xe, src_ids, Bt1);
    conv_wt_kernel<<<(DD * DD) / T, T, 0, stream>>>(W1, W1t);
    conv_wt_kernel<<<(DD * DD) / T, T, 0, stream>>>(W2, W2t);
    qproj_kernel<<<(N_ALLC * DQC) / T, T, 0, stream>>>(Xe, src_ids, Wdis, q);

    // 4) attention scores -> dense A[6144x384] (f32 atomics), then f16 copy
    attn_edge_kernel<<<E_TGC / T, T, 0, stream>>>(q, tg_src, tg_dst, Adense, Asum, Rsum);
    denom_kernel<<<N_DSTC / T, T, 0, stream>>>(Asum, Rsum, invden);
    conv_f16_kernel<<<(N_DSTC * N_SRCC) / T, T, 0, stream>>>(Adense, Ah, N_DSTC * N_SRCC);

    // 5) GEMM1 (WMMA): Xe_trans = (Adense @ SRC + Xe*Rsum) * invden, + BN1 stats
    {
        int tiles = (N_DSTC / 16) * (DD / 16);            // 6144 waves
        wmma_gemm_kernel<<<tiles / 8, T, 0, stream>>>(
            Ah, Bt1, Xetr, N_DSTC, N_SRCC, /*mode=*/0,
            Xe, Rsum, invden, nullptr, cs1, cq1);
    }
    bn_finalize_kernel<<<1, DD, 0, stream>>>(cs1, cq1, mean1, istd1);
    ew_xe2_kernel<<<(N_DSTC * DD) / T, T, 0, stream>>>(Xe, Xetr, mean1, istd1,
                                                       bn1g, bn1b, Xe2, Xe2h);

    // 6) GEMM2 (WMMA): Z = Xe2 @ W1 + b1, + BN2 stats; h16 = relu(BN2(Z))
    {
        int tiles = (N_DSTC / 16) * (DD / 16);
        wmma_gemm_kernel<<<tiles / 8, T, 0, stream>>>(
            Xe2h, W1t, Z, N_DSTC, DD, /*mode=*/1,
            nullptr, nullptr, nullptr, b1, cs2, cq2);
    }
    bn_finalize_kernel<<<1, DD, 0, stream>>>(cs2, cq2, mean2, istd2);
    ew_h_kernel<<<(N_DSTC * DD) / T, T, 0, stream>>>(Z, mean2, istd2, bn2g, bn2b, h16);

    // 7) GEMM3 (WMMA): Z = h @ W2 + b2, + BN3 stats; out = Xe2 + BN3(Z)
    {
        int tiles = (N_DSTC / 16) * (DD / 16);
        wmma_gemm_kernel<<<tiles / 8, T, 0, stream>>>(
            h16, W2t, Z, N_DSTC, DD, /*mode=*/1,
            nullptr, nullptr, nullptr, b2, cs3, cq3);
    }
    bn_finalize_kernel<<<1, DD, 0, stream>>>(cs3, cq3, mean3, istd3);
    ew_out_kernel<<<(N_DSTC * DD) / T, T, 0, stream>>>(Xe2, Z, mean3, istd3, bn3g, bn3b, out);
}
